// MaskedAttention_7095285973109
// MI455X (gfx1250) — matrix operand order repaired and checked by reading, not compiled, not run on hardware
//
#include <hip/hip_runtime.h>

// ---------------------------------------------------------------------------
// Masked multi-head attention forward for gfx1250 (MI455X), wave32 + WMMA.
// Pipeline (3 kernels):
//   1) qkv gemm (f16 WMMA, W transposed through LDS)
//   2) fused attention: online-softmax stats pass (register-only) +
//      recompute pass writing probs and accumulating O = P@V
//   3) output projection gemm + bias
// Attn tensor touches HBM exactly once (the mandatory 210MB probs write).
// ---------------------------------------------------------------------------

typedef __attribute__((ext_vector_type(16))) _Float16 v16h;
typedef __attribute__((ext_vector_type(8)))  _Float16 v8h;
typedef __attribute__((ext_vector_type(8)))  float    v8f;

#define BATCH  4
#define NSEQ   1280
#define DIMC   512
#define HEADS  8
#define HD     64
#define NGRP   5        // NUM_CLASS + 1
#define GRPW   256      // NSEQ / NGRP

#define NEG_INF (-__builtin_inff())

// ---- WMMA fragment helpers (layouts per CDNA5 ISA 7.12.2, wave32) ---------

// A fragment, 16x32 f16, source row-major (ld in halves).
// lane<16 : row=lane,     halves[0..7]=K 0..7,  halves[8..15]=K 16..23
// lane>=16: row=lane-16,  halves[0..7]=K 8..15, halves[8..15]=K 24..31
__device__ __forceinline__ v16h loadA16(const _Float16* __restrict__ base, int ld) {
    const int lane = threadIdx.x & 31;
    const int row  = lane & 15;
    const int sel  = lane >> 4;
    const _Float16* p = base + row * ld + sel * 8;
    v8h lo = *(const v8h*)(p);
    v8h hi = *(const v8h*)(p + 16);
    v16h r;
#pragma unroll
    for (int i = 0; i < 8; ++i) { r[i] = lo[i]; r[i + 8] = hi[i]; }
    return r;
}

// A fragment from f32 source, converted to f16 in-register.
__device__ __forceinline__ v16h loadA32cvt(const float* __restrict__ base, int ld) {
    const int lane = threadIdx.x & 31;
    const int row  = lane & 15;
    const int sel  = lane >> 4;
    const float* p = base + row * ld + sel * 8;
    v16h r;
#pragma unroll
    for (int i = 0; i < 8; ++i) {
        r[i]     = (_Float16)p[i];
        r[i + 8] = (_Float16)p[16 + i];
    }
    return r;
}

// B fragment, 32x16 f16, source N-major ("transposed": column n contiguous in K,
// stride ldt halves).  lane<16: col=lane, K=0..7,16..23 ; lane>=16: col=lane-16, K=8..15,24..31
__device__ __forceinline__ v16h loadB16(const _Float16* __restrict__ baseT, int ldt) {
    const int lane = threadIdx.x & 31;
    const int col  = lane & 15;
    const int sel  = lane >> 4;
    const _Float16* p = baseT + col * ldt + sel * 8;
    v8h lo = *(const v8h*)(p);
    v8h hi = *(const v8h*)(p + 16);
    v16h r;
#pragma unroll
    for (int i = 0; i < 8; ++i) { r[i] = lo[i]; r[i + 8] = hi[i]; }
    return r;
}

__device__ __forceinline__ v8f wmma16(v16h a, v16h b, v8f c) {
    return __builtin_amdgcn_wmma_f32_16x16x32_f16(false, a, false, b,
                                                  (short)0, c, false, false);
}

// ---------------------------------------------------------------------------
// Kernel 1: qkv = x(5120x512) @ Wqkv(512x1536); scatter to q/k/vT f16 buffers.
//   grid (80,12), block 256.  Wave w: rows m0+(w&3)*16, cols n0+(w>>2)*64.
// ---------------------------------------------------------------------------
__global__ __launch_bounds__(256)
void qkv_gemm_kernel(const float* __restrict__ x, const float* __restrict__ Wqkv,
                     _Float16* __restrict__ qbuf, _Float16* __restrict__ kbuf,
                     _Float16* __restrict__ vTbuf) {
    __shared__ __align__(16) _Float16 sB[128 * 40];   // W tile, N-major, padded
    const int tid = threadIdx.x;
    const int m0  = blockIdx.x * 64;
    const int n0  = blockIdx.y * 128;
    const int w   = tid >> 5;
    const int mw  = (w & 3) * 16;
    const int nw  = (w >> 2) * 64;

    v8f acc[4] = {};

    for (int k0 = 0; k0 < DIMC; k0 += 32) {
        // stage W[k0..k0+32][n0..n0+128] transposed -> sB[n*40 + k]
        for (int idx = tid; idx < 32 * 128; idx += 256) {
            int k = idx >> 7, n = idx & 127;
            sB[n * 40 + k] = (_Float16)Wqkv[(size_t)(k0 + k) * (3 * DIMC) + n0 + n];
        }
        __syncthreads();
        v16h a = loadA32cvt(x + (size_t)(m0 + mw) * DIMC + k0, DIMC);
#pragma unroll
        for (int j = 0; j < 4; ++j) {
            v16h b = loadB16(sB + (nw + j * 16) * 40, 40);
            acc[j] = wmma16(a, b, acc[j]);
        }
        __syncthreads();
    }

    const int lane = tid & 31;
    const int sel  = lane >> 4;
    const int cn   = lane & 15;
#pragma unroll
    for (int j = 0; j < 4; ++j) {
#pragma unroll
        for (int r = 0; r < 8; ++r) {
            const int gm = m0 + mw + r + sel * 8;
            const int gn = n0 + nw + j * 16 + cn;
            const int b  = gm / NSEQ, n = gm % NSEQ;
            const int which = gn >> 9;          // j / 512
            const int h     = (gn >> 6) & 7;    // (j % 512) / 64
            const int d     = gn & 63;
            const int bh    = b * HEADS + h;
            const _Float16 val = (_Float16)acc[j][r];
            if (which == 0)      qbuf[((size_t)bh * NSEQ + n) * HD + d] = val;
            else if (which == 1) kbuf[((size_t)bh * NSEQ + n) * HD + d] = val;
            else                 vTbuf[((size_t)bh * HD + d) * NSEQ + n] = val;
        }
    }
}

// ---------------------------------------------------------------------------
// Kernel 2 (fused): per head, each wave owns 16 query rows end-to-end.
//  Pass 1: S tiles via WMMA kept in registers; online per-row max / exp-sum
//          (per-lane partials + half-wave shfl_xor reduction). No memory spill.
//  Pass 2: recompute S tiles (WMMA is cheaper than a 420MB round-trip),
//          p = exp(v-M)/s written to d_out (the only attn HBM pass),
//          P panel bounced through wave-private LDS (C-layout -> A-layout),
//          O += P@V with WMMA; O stored f16 in (B,N,C) layout.
//  grid (10, 32), block 256 (8 independent waves).
// ---------------------------------------------------------------------------
__global__ __launch_bounds__(256)
void fused_attn_kernel(const _Float16* __restrict__ qbuf,
                       const _Float16* __restrict__ kbuf,
                       const _Float16* __restrict__ vTbuf,
                       const int* __restrict__ mask,
                       float* __restrict__ attn,
                       _Float16* __restrict__ obuf) {
    __shared__ __align__(16) _Float16 sP[8][16 * 40];  // wave-private P panels
    const int tid  = threadIdx.x;
    const int w    = tid >> 5;
    const int bh   = blockIdx.y;
    const int b    = bh >> 3;
    const int h    = bh & 7;
    const int m0   = blockIdx.x * 128 + w * 16;
    const int lane = tid & 31;
    const int sel  = lane >> 4;
    const int cn   = lane & 15;

    const _Float16* qbase = qbuf + ((size_t)bh * NSEQ + m0) * HD;
    const _Float16* kbase = kbuf + (size_t)bh * NSEQ * HD;
    const _Float16* vT    = vTbuf + (size_t)bh * HD * NSEQ;
    float* arow = attn + (size_t)bh * NSEQ * NSEQ;
    _Float16* sPw = &sP[w][0];

    // allowed-group bitmask: groups 0..3 from mask, group 4 always allowed
    int abits = 1 << 4;
#pragma unroll
    for (int g = 0; g < 4; ++g) abits |= (mask[b * 4 + g] != 0) << g;

    const v16h a0 = loadA16(qbase, HD);        // d = 0..31
    const v16h a1 = loadA16(qbase + 32, HD);   // d = 32..63
    const float scale = 0.125f;                // hd^-0.5

    // ---- pass 1: online row max + rescaled exp-sum (per-lane partials) ----
    float ml[8], ls[8];
#pragma unroll
    for (int r = 0; r < 8; ++r) { ml[r] = NEG_INF; ls[r] = 0.f; }

    for (int n0 = 0; n0 < NSEQ; n0 += 16) {
        if (n0 + 16 < NSEQ)
            __builtin_prefetch(kbase + (size_t)(n0 + 16) * HD + (size_t)cn * HD, 0, 1);
        v16h b0 = loadB16(kbase + (size_t)n0 * HD, HD);
        v16h b1 = loadB16(kbase + (size_t)n0 * HD + 32, HD);
        v8f acc = {};
        acc = wmma16(a0, b0, acc);
        acc = wmma16(a1, b1, acc);
        const bool okc = (abits >> ((n0 + cn) / GRPW)) & 1;
#pragma unroll
        for (int r = 0; r < 8; ++r) {
            const float v  = okc ? acc[r] * scale : NEG_INF;
            const float mn = fmaxf(ml[r], v);
            const float f  = (mn == ml[r]) ? 1.f : __expf(ml[r] - mn); // -inf safe
            const float e  = (v == NEG_INF) ? 0.f : __expf(v - mn);
            ls[r] = ls[r] * f + e;
            ml[r] = mn;
        }
    }
    // combine the 16 column-lanes of each half (rows r / 8+r)
#pragma unroll
    for (int r = 0; r < 8; ++r) {
        float M = ml[r];
#pragma unroll
        for (int o = 1; o < 16; o <<= 1) M = fmaxf(M, __shfl_xor(M, o, 32));
        float c = (ml[r] == NEG_INF) ? 0.f : ls[r] * __expf(ml[r] - M);
#pragma unroll
        for (int o = 1; o < 16; o <<= 1) c += __shfl_xor(c, o, 32);
        ml[r] = M;          // row max
        ls[r] = 1.0f / c;   // row inv-sum (group 4 always allowed => c > 0)
    }

    // ---- pass 2: recompute S, emit probs, accumulate O = P @ V ------------
    v8f accO[4] = {};
    for (int n0 = 0; n0 < NSEQ; n0 += 32) {
#pragma unroll
        for (int t = 0; t < 2; ++t) {
            const int nt = n0 + t * 16;
            v16h b0 = loadB16(kbase + (size_t)nt * HD, HD);
            v16h b1 = loadB16(kbase + (size_t)nt * HD + 32, HD);
            v8f acc = {};
            acc = wmma16(a0, b0, acc);
            acc = wmma16(a1, b1, acc);
            const bool okc = (abits >> ((nt + cn) / GRPW)) & 1;
#pragma unroll
            for (int r = 0; r < 8; ++r) {
                const float v = okc ? acc[r] * scale : NEG_INF;
                const float p = (v == NEG_INF) ? 0.f : __expf(v - ml[r]) * ls[r];
                arow[(size_t)(m0 + r + sel * 8) * NSEQ + nt + cn] = p;       // attn out
                sPw[(r + sel * 8) * 40 + t * 16 + cn] = (_Float16)p;         // transpose
            }
        }
        v16h ap = loadA16(sPw, 40);   // P fragment, cols n0..n0+31 (wave-private LDS)
#pragma unroll
        for (int j = 0; j < 4; ++j) { // d tiles 0..15,16..31,32..47,48..63
            v16h bf = loadB16(vT + (size_t)(j * 16) * NSEQ + n0, NSEQ);
            accO[j] = wmma16(ap, bf, accO[j]);
        }
    }

#pragma unroll
    for (int j = 0; j < 4; ++j) {
#pragma unroll
        for (int r = 0; r < 8; ++r) {
            const int gm = m0 + r + sel * 8;
            obuf[((size_t)b * NSEQ + gm) * DIMC + h * HD + j * 16 + cn] = (_Float16)accO[j][r];
        }
    }
}

// ---------------------------------------------------------------------------
// Kernel 3: out = O(5120x512 f16) @ Wproj(512x512) + bproj  -> f32.
//   grid (80,4), block 256; Wproj transposed through LDS like kernel 1.
// ---------------------------------------------------------------------------
__global__ __launch_bounds__(256)
void proj_gemm_kernel(const _Float16* __restrict__ obuf,
                      const float* __restrict__ Wproj,
                      const float* __restrict__ bproj,
                      float* __restrict__ out) {
    __shared__ __align__(16) _Float16 sB[128 * 40];
    const int tid = threadIdx.x;
    const int m0  = blockIdx.x * 64;
    const int n0  = blockIdx.y * 128;
    const int w   = tid >> 5;
    const int mw  = (w & 3) * 16;
    const int nw  = (w >> 2) * 64;

    v8f acc[4] = {};
    for (int k0 = 0; k0 < DIMC; k0 += 32) {
        for (int idx = tid; idx < 32 * 128; idx += 256) {
            int k = idx >> 7, n = idx & 127;
            sB[n * 40 + k] = (_Float16)Wproj[(size_t)(k0 + k) * DIMC + n0 + n];
        }
        __syncthreads();
        v16h a = loadA16(obuf + (size_t)(m0 + mw) * DIMC + k0, DIMC);
#pragma unroll
        for (int j = 0; j < 4; ++j) {
            v16h b = loadB16(sB + (nw + j * 16) * 40, 40);
            acc[j] = wmma16(a, b, acc[j]);
        }
        __syncthreads();
    }

    const int lane = tid & 31;
    const int sel  = lane >> 4;
    const int cn   = lane & 15;
#pragma unroll
    for (int j = 0; j < 4; ++j) {
#pragma unroll
        for (int r = 0; r < 8; ++r) {
            const int gm = m0 + mw + r + sel * 8;
            const int gn = n0 + nw + j * 16 + cn;
            out[(size_t)gm * DIMC + gn] = acc[j][r] + bproj[gn];
        }
    }
}

// ---------------------------------------------------------------------------
extern "C" void kernel_launch(void* const* d_in, const int* in_sizes, int n_in,
                              void* d_out, int out_size, void* d_ws, size_t ws_size,
                              hipStream_t stream) {
    const float* x     = (const float*)d_in[0];
    const int*   mask  = (const int*)d_in[1];
    const float* Wqkv  = (const float*)d_in[2];
    const float* Wproj = (const float*)d_in[3];
    const float* bproj = (const float*)d_in[4];

    float* out  = (float*)d_out;
    float* attn = out + (size_t)BATCH * NSEQ * DIMC;          // 2,621,440 offset

    const size_t headElems = (size_t)BATCH * HEADS * NSEQ * HD;  // 2,621,440
    _Float16* qbuf = (_Float16*)d_ws;
    _Float16* kbuf = qbuf + headElems;
    _Float16* vT   = kbuf + headElems;
    _Float16* obuf = vT + headElems;          // total ws use: ~21 MB (f16)

    qkv_gemm_kernel<<<dim3(80, 12), 256, 0, stream>>>(x, Wqkv, qbuf, kbuf, vT);
    fused_attn_kernel<<<dim3(10, 32), 256, 0, stream>>>(qbuf, kbuf, vT, mask, attn, obuf);
    proj_gemm_kernel<<<dim3(80, 4), 256, 0, stream>>>(obuf, Wproj, bproj, out);
}
